// pointnet_fp_module_2697239462399
// MI455X (gfx1250) — matrix operand order repaired and checked by reading, not compiled, not run on hardware
//
#include <hip/hip_runtime.h>

// ---------------- types / constants ----------------
typedef __bf16 bf16;
typedef __attribute__((ext_vector_type(16))) __bf16 v16bf;
typedef __attribute__((ext_vector_type(8)))  __bf16 v8bf;
typedef __attribute__((ext_vector_type(8)))  float  v8f;

#define BN_EPS 1e-5f

static constexpr int BATCH = 8;
static constexpr int N1    = 16384;
static constexpr int N2    = 1024;
static constexpr int C1    = 128;
static constexpr int C2    = 256;
static constexpr int CIN   = 384;            // C2 + C1
static constexpr int HOUT  = 256;            // both MLP widths
static constexpr int MROWS = BATCH * N1;     // 131072
static constexpr int MT    = MROWS / 64;     // 2048 M-tiles

// LDS row pitch (bf16 elems): 40 -> 80B -> 20-bank stride, conflict-free b128 frag loads
static constexpr int PITCH = 40;

// ---------------- kernel 1: weight transpose + bf16 convert ----------------
// Wt[n*K + k] = (bf16) W[k*N + n]   (K-major per output column: WMMA-B friendly)
__global__ void __launch_bounds__(256) prep_w_kernel(const float* __restrict__ W,
                                                     bf16* __restrict__ Wt,
                                                     int K, int N) {
  int idx = blockIdx.x * 256 + threadIdx.x;      // grid sized exactly K*N/256
  int n = idx / K, k = idx - n * K;
  Wt[idx] = (bf16)W[(size_t)k * N + n];
}

// ---------------- kernel 2: three_nn + interpolate + concat -> x bf16 ----------------
__global__ void __launch_bounds__(256) fp_interp_kernel(const float* __restrict__ xyz1,
                                                        const float* __restrict__ xyz2,
                                                        const float* __restrict__ points1,
                                                        const float* __restrict__ points2,
                                                        bf16* __restrict__ xout) {
  __shared__ float sx[N2], sy[N2], sz[N2];
  __shared__ int   si0[256], si1[256], si2[256];
  __shared__ float sw0[256], sw1[256], sw2[256];

  const int t = threadIdx.x;
  const int b = blockIdx.y;
  const int rowbase = blockIdx.x * 256;

  const float* x2 = xyz2 + (size_t)b * (N2 * 3);
  for (int i = t; i < N2 * 3; i += 256) {
    float v = x2[i];
    int p = i / 3, c = i - 3 * p;
    (c == 0 ? sx : (c == 1 ? sy : sz))[p] = v;
  }
  __syncthreads();

  const int n = rowbase + t;
  const float* pc = xyz1 + ((size_t)b * N1 + n) * 3;
  float px = pc[0], py = pc[1], pz = pc[2];
  float d0 = 3.4e38f, d1 = 3.4e38f, d2 = 3.4e38f;
  int   i0 = 0, i1 = 0, i2 = 0;
  for (int j = 0; j < N2; ++j) {
    float dx = px - sx[j], dy = py - sy[j], dz = pz - sz[j];
    float dd = dx * dx + dy * dy + dz * dz;
    if (dd < d2) {
      if (dd < d1) {
        if (dd < d0) { d2 = d1; i2 = i1; d1 = d0; i1 = i0; d0 = dd; i0 = j; }
        else         { d2 = d1; i2 = i1; d1 = dd; i1 = j; }
      } else         { d2 = dd; i2 = j; }
    }
  }
  float r0 = 1.f / fmaxf(sqrtf(fmaxf(d0, 0.f)), 1e-10f);
  float r1 = 1.f / fmaxf(sqrtf(fmaxf(d1, 0.f)), 1e-10f);
  float r2 = 1.f / fmaxf(sqrtf(fmaxf(d2, 0.f)), 1e-10f);
  float rs = 1.f / (r0 + r1 + r2);
  si0[t] = i0; si1[t] = i1; si2[t] = i2;
  sw0[t] = r0 * rs; sw1[t] = r1 * rs; sw2[t] = r2 * rs;
  __syncthreads();

  const float* p2b = points2 + (size_t)b * N2 * C2;
  for (int p = 0; p < 256; ++p) {
    size_t m = (size_t)b * N1 + rowbase + p;
    float v = sw0[p] * p2b[(size_t)si0[p] * C2 + t]
            + sw1[p] * p2b[(size_t)si1[p] * C2 + t]
            + sw2[p] * p2b[(size_t)si2[p] * C2 + t];
    bf16* xr = xout + m * CIN;
    xr[t] = (bf16)v;
    if (t < C1) xr[C2 + t] = (bf16)points1[m * C1 + t];
  }
}

// ---------------- kernel 3/5: WMMA GEMM + bias + BN partial stats ----------------
// Block tile 64 x 256 (full N): A read exactly once from HBM. 8 waves = 4(M) x 2(N),
// each wave 16 x 128 via 8x v_wmma_f32_16x16x32_bf16. Double-buffered LDS fed by
// global_load_async_to_lds_b128 (ASYNCcnt pipeline). FUSE: A = relu(y*a+s) built in
// VALU (bf16 convert) and ds_store'd; B stays on the async path.
template <int KDIM, bool FUSE>
__global__ void __launch_bounds__(256) gemm_bn_kernel(const bf16*  __restrict__ Abf,
                                                      const float* __restrict__ Afp,
                                                      const float* __restrict__ bn_a,
                                                      const float* __restrict__ bn_s,
                                                      const bf16*  __restrict__ Wt,
                                                      const float* __restrict__ bias,
                                                      float* __restrict__ Cout,
                                                      float* __restrict__ psum,
                                                      float* __restrict__ psumsq) {
  __shared__ bf16  Alds[2][64 * PITCH];     // 64 x 32 per buffer
  __shared__ bf16  Blds[2][256 * PITCH];    // 256 cols x 32 K per buffer (K-major)
  __shared__ float sSum[256], sSq[256];
  __shared__ float sA[256], sS[256];

  const int t    = threadIdx.x;
  const int bm   = blockIdx.x;          // 64-row tile
  const int lane = t & 31;
  const int wave = t >> 5;
  const int wm   = wave & 3;            // 4 waves along M
  const int wn   = wave >> 2;           // 2 waves along N (128 cols each)
  const int l15  = lane & 15;
  const int hi   = lane >> 4;

  sSum[t] = 0.f; sSq[t] = 0.f;
  if constexpr (FUSE) { sA[t] = bn_a[t]; sS[t] = bn_s[t]; }
  __syncthreads();

  v8f acc[8] = {};

  const int    arow  = t >> 2, ac8 = (t & 3) * 8;       // A staging: 16B/thread
  const size_t aElem = (size_t)(bm * 64 + arow) * KDIM + ac8;

  // ---- staging: issue async copies (B always; A when !FUSE), ds_store A when FUSE
  auto stage = [&](int kt, int buf) {
    // B: thread t = column; 64B (4 x b128), instr offset applies to LDS and global.
    {
      unsigned ldsB = (unsigned)(size_t)&Blds[buf][t * PITCH];
      unsigned gofB = (unsigned)(((size_t)t * KDIM + kt * 32) * sizeof(bf16));
      asm volatile("global_load_async_to_lds_b128 %0, %1, %2 offset:0\n\t"
                   "global_load_async_to_lds_b128 %0, %1, %2 offset:16\n\t"
                   "global_load_async_to_lds_b128 %0, %1, %2 offset:32\n\t"
                   "global_load_async_to_lds_b128 %0, %1, %2 offset:48"
                   :: "v"(ldsB), "v"(gofB), "s"(Wt) : "memory");
    }
    if constexpr (!FUSE) {
      unsigned ldsA = (unsigned)(size_t)&Alds[buf][arow * PITCH + ac8];
      unsigned gofA = (unsigned)((aElem + (size_t)kt * 32) * sizeof(bf16));
      asm volatile("global_load_async_to_lds_b128 %0, %1, %2"
                   :: "v"(ldsA), "v"(gofA), "s"(Abf) : "memory");
    } else {
      const float4* ap = (const float4*)(Afp + aElem + (size_t)kt * 32);
      float4 f0 = ap[0], f1 = ap[1];
      union { int4 i4v; bf16 h[8]; } pk;
      int ch = kt * 32 + ac8;
      pk.h[0] = (bf16)fmaxf(0.f, f0.x * sA[ch + 0] + sS[ch + 0]);
      pk.h[1] = (bf16)fmaxf(0.f, f0.y * sA[ch + 1] + sS[ch + 1]);
      pk.h[2] = (bf16)fmaxf(0.f, f0.z * sA[ch + 2] + sS[ch + 2]);
      pk.h[3] = (bf16)fmaxf(0.f, f0.w * sA[ch + 3] + sS[ch + 3]);
      pk.h[4] = (bf16)fmaxf(0.f, f1.x * sA[ch + 4] + sS[ch + 4]);
      pk.h[5] = (bf16)fmaxf(0.f, f1.y * sA[ch + 5] + sS[ch + 5]);
      pk.h[6] = (bf16)fmaxf(0.f, f1.z * sA[ch + 6] + sS[ch + 6]);
      pk.h[7] = (bf16)fmaxf(0.f, f1.w * sA[ch + 7] + sS[ch + 7]);
      *(int4*)&Alds[buf][arow * PITCH + ac8] = pk.i4v;
    }
  };

  // ---- compute one K-step (32) from LDS buffer `buf`
  auto compute = [&](int buf) {
    // A frag (ISA 7.12.2): lane<16 -> K0..7,K16..23 ; lane>=16 -> K8..15,K24..31
    union { v16bf v; v8bf h[2]; } af;
    const bf16* ab = &Alds[buf][(wm * 16 + l15) * PITCH + hi * 8];
    af.h[0] = *(const v8bf*)(ab);
    af.h[1] = *(const v8bf*)(ab + 16);
#pragma unroll
    for (int j = 0; j < 8; ++j) {
      union { v16bf v; v8bf h[2]; } bfr;
      const bf16* bb = &Blds[buf][(wn * 128 + j * 16 + l15) * PITCH + hi * 8];
      bfr.h[0] = *(const v8bf*)(bb);
      bfr.h[1] = *(const v8bf*)(bb + 16);
      acc[j] = __builtin_amdgcn_wmma_f32_16x16x32_bf16(
          false, af.v, false, bfr.v, (short)0, acc[j], false, false);
    }
  };

  constexpr int T = KDIM / 32;
  stage(0, 0);
  for (int kt = 0; kt < T - 1; ++kt) {
    stage(kt + 1, (kt + 1) & 1);
    // wait for tile kt's async copies (in-order): next tile leaves 5 (A+B) / 4 (B) outstanding
    if constexpr (FUSE) asm volatile("s_wait_asynccnt 0x4" ::: "memory");
    else                asm volatile("s_wait_asynccnt 0x5" ::: "memory");
    __syncthreads();
    compute(kt & 1);
    __syncthreads();   // all waves done reading buf before tile kt+2 overwrites it
  }
  asm volatile("s_wait_asynccnt 0x0" ::: "memory");
  __syncthreads();
  compute((T - 1) & 1);

  // epilogue: bias, fp32 store, per-channel partial sums for BN
  const int rowb = bm * 64 + wm * 16 + hi * 8;
#pragma unroll
  for (int j = 0; j < 8; ++j) {
    const int gcol = wn * 128 + j * 16 + l15;
    float bv = bias[gcol];
    float s = 0.f, q = 0.f;
#pragma unroll
    for (int r = 0; r < 8; ++r) {
      float v = acc[j][r] + bv;
      Cout[(size_t)(rowb + r) * HOUT + gcol] = v;
      s += v; q += v * v;
    }
    atomicAdd(&sSum[gcol], s);
    atomicAdd(&sSq[gcol], q);
  }
  __syncthreads();
  psum  [(size_t)bm * HOUT + t] = sSum[t];
  psumsq[(size_t)bm * HOUT + t] = sSq[t];
}

// ---------------- kernel 4/6: BN stats reduce -> scale/shift ----------------
__global__ void __launch_bounds__(256) bn_stats_kernel(const float* __restrict__ psum,
                                                       const float* __restrict__ psumsq,
                                                       const float* __restrict__ g,
                                                       const float* __restrict__ beta,
                                                       float* __restrict__ a_out,
                                                       float* __restrict__ s_out,
                                                       int ntiles, float invM) {
  __shared__ float rs[256], rq[256];
  const int c = blockIdx.x;
  const int j = threadIdx.x;
  float s = 0.f, q = 0.f;
  for (int tdx = j; tdx < ntiles; tdx += 256) {
    s += psum  [(size_t)tdx * HOUT + c];
    q += psumsq[(size_t)tdx * HOUT + c];
  }
  rs[j] = s; rq[j] = q;
  __syncthreads();
  for (int st = 128; st > 0; st >>= 1) {
    if (j < st) { rs[j] += rs[j + st]; rq[j] += rq[j + st]; }
    __syncthreads();
  }
  if (j == 0) {
    float mean = rs[0] * invM;
    float var  = rq[0] * invM - mean * mean;
    float a    = g[c] * rsqrtf(var + BN_EPS);
    a_out[c] = a;
    s_out[c] = beta[c] - mean * a;
  }
}

// ---------------- kernel 7: final BN+ReLU in place on d_out ----------------
__global__ void __launch_bounds__(256) bn_relu_out_kernel(float* __restrict__ out,
                                                          const float* __restrict__ a,
                                                          const float* __restrict__ s) {
  size_t i4 = ((size_t)blockIdx.x * 256 + threadIdx.x) * 4;
  float4* p = (float4*)(out + i4);
  float4 v = *p;
  int c = (int)(i4 & (HOUT - 1));
  v.x = fmaxf(0.f, v.x * a[c + 0] + s[c + 0]);
  v.y = fmaxf(0.f, v.y * a[c + 1] + s[c + 1]);
  v.z = fmaxf(0.f, v.z * a[c + 2] + s[c + 2]);
  v.w = fmaxf(0.f, v.w * a[c + 3] + s[c + 3]);
  *p = v;
}

// ---------------- launch ----------------
extern "C" void kernel_launch(void* const* d_in, const int* in_sizes, int n_in,
                              void* d_out, int out_size, void* d_ws, size_t ws_size,
                              hipStream_t stream) {
  (void)in_sizes; (void)n_in; (void)out_size; (void)ws_size;
  const float* xyz1    = (const float*)d_in[0];
  const float* xyz2    = (const float*)d_in[1];
  const float* points1 = (const float*)d_in[2];
  const float* points2 = (const float*)d_in[3];
  const float* W0  = (const float*)d_in[4];
  const float* b0  = (const float*)d_in[5];
  const float* g0  = (const float*)d_in[6];
  const float* be0 = (const float*)d_in[7];
  const float* W1  = (const float*)d_in[8];
  const float* b1  = (const float*)d_in[9];
  const float* g1  = (const float*)d_in[10];
  const float* be1 = (const float*)d_in[11];
  float* out = (float*)d_out;

  char* w = (char*)d_ws;
  bf16*  x_bf = (bf16*)w;   w += (size_t)MROWS * CIN * sizeof(bf16);   // 100.7 MB
  bf16*  wt0  = (bf16*)w;   w += (size_t)HOUT * CIN * sizeof(bf16);    // 192 KB
  bf16*  wt1  = (bf16*)w;   w += (size_t)HOUT * HOUT * sizeof(bf16);   // 128 KB
  float* y1   = (float*)w;  w += (size_t)MROWS * HOUT * sizeof(float); // 134 MB
  float* ps1  = (float*)w;  w += (size_t)MT * HOUT * sizeof(float);
  float* pq1  = (float*)w;  w += (size_t)MT * HOUT * sizeof(float);
  float* ps2  = (float*)w;  w += (size_t)MT * HOUT * sizeof(float);
  float* pq2  = (float*)w;  w += (size_t)MT * HOUT * sizeof(float);
  float* a1   = (float*)w;  w += HOUT * sizeof(float);
  float* s1   = (float*)w;  w += HOUT * sizeof(float);
  float* a2   = (float*)w;  w += HOUT * sizeof(float);
  float* s2   = (float*)w;  w += HOUT * sizeof(float);

  const float invM = 1.0f / (float)MROWS;

  prep_w_kernel<<<(CIN * HOUT) / 256, 256, 0, stream>>>(W0, wt0, CIN, HOUT);
  prep_w_kernel<<<(HOUT * HOUT) / 256, 256, 0, stream>>>(W1, wt1, HOUT, HOUT);

  fp_interp_kernel<<<dim3(N1 / 256, BATCH), 256, 0, stream>>>(xyz1, xyz2, points1, points2, x_bf);

  gemm_bn_kernel<CIN, false><<<MT, 256, 0, stream>>>(
      x_bf, nullptr, nullptr, nullptr, wt0, b0, y1, ps1, pq1);

  bn_stats_kernel<<<HOUT, 256, 0, stream>>>(ps1, pq1, g0, be0, a1, s1, MT, invM);

  gemm_bn_kernel<HOUT, true><<<MT, 256, 0, stream>>>(
      nullptr, y1, a1, s1, wt1, b1, out, ps2, pq2);

  bn_stats_kernel<<<HOUT, 256, 0, stream>>>(ps2, pq2, g1, be1, a2, s2, MT, invM);

  bn_relu_out_kernel<<<((size_t)MROWS * HOUT) / 1024, 256, 0, stream>>>(out, a2, s2);
}
